// CrossAttention_90211493085731
// MI455X (gfx1250) — compile-verified, matrix operand order checked by reading, not run on hardware
//
#include <hip/hip_runtime.h>

#define LL 4
#define NN 2048
#define DIN 1024
#define HH 8
#define DHH 64
#define INNERD 512
#define ATTN_SCALE 0.125f
#define NEGMAX -3.402823466e38f

typedef __attribute__((ext_vector_type(16))) __bf16 v16bf;
typedef __attribute__((ext_vector_type(8)))  float  v8f;
typedef __attribute__((ext_vector_type(4)))  unsigned int u32x4;
typedef __attribute__((ext_vector_type(8)))  int i32x8;
typedef __attribute__((ext_vector_type(4)))  int i32x4;

union Frag16 {
  uint4 u[2];
  unsigned short s[16];
  v16bf v;
};

// native BF16 convert (round-to-nearest-even fptrunc -> v_cvt_*bf16_f32)
__device__ __forceinline__ unsigned short f2bf(float f) {
  __bf16 h = (__bf16)f;
  return __builtin_bit_cast(unsigned short, h);
}

// A/B fragment gather: elements 0-7 = p[0..7], 8-15 = p[16..23]
__device__ __forceinline__ v16bf frag_ld_u16(const unsigned short* p) {
  Frag16 f;
  f.u[0] = *(const uint4*)(p);
  f.u[1] = *(const uint4*)(p + 16);
  return f.v;
}

__device__ __forceinline__ v16bf frag_ld_f32(const float* p) {
  Frag16 f;
#pragma unroll
  for (int i = 0; i < 8; ++i) f.s[i] = f2bf(p[i]);
#pragma unroll
  for (int i = 0; i < 8; ++i) f.s[8 + i] = f2bf(p[16 + i]);
  return f.v;
}

__device__ __forceinline__ v8f wmma_bf16(v16bf a, v16bf b, v8f c) {
  return __builtin_amdgcn_wmma_f32_16x16x32_bf16(false, a, false, b,
                                                 (short)0, c, false, false);
}

// ---------------------------------------------------------------------------
// TDM: DMA a 2-D tile (16-bit elements) Global -> LDS via tensor descriptor.
// td0/td1 = tensor dims (elems), tile0/tile1 = tile dims, stride0 = row stride.
// D# bit layout per CDNA5 ISA ch.8 (group0 128b, group1 256b; groups 2/3 zero).
// 6-arg builtin form (amdgpu-toolchain clang-23 / therock-10.0 headers).
// ---------------------------------------------------------------------------
__device__ __forceinline__ void tdm_load_2d_b16(unsigned lds_off, const void* gptr,
                                                unsigned td0, unsigned td1,
                                                unsigned tile0, unsigned tile1,
                                                unsigned stride0) {
  unsigned long long ga = (unsigned long long)(size_t)gptr;
  u32x4 g0;
  g0.x = 1u;                                          // count=1, user mode
  g0.y = lds_off;                                     // lds_addr (bytes)
  g0.z = (unsigned)ga;                                // global_addr[31:0]
  g0.w = (unsigned)((ga >> 32) & 0x1FFFFFFu) | (2u << 30);  // addr[56:32], type=2
  i32x8 g1;
  g1[0] = (int)(1u << 16);                            // data_size = 2 bytes
  g1[1] = (int)((td0 & 0xFFFFu) << 16);               // [47:32]=0, td0 lo16
  g1[2] = (int)((td0 >> 16) | ((td1 & 0xFFFFu) << 16));
  g1[3] = (int)((td1 >> 16) | (tile0 << 16));
  g1[4] = (int)tile1;                                 // tile_dim2 = 0 (2-D)
  g1[5] = (int)stride0;                               // dim0 stride [31:0]
  g1[6] = 0;                                          // stride0 hi / dim1 stride lo
  g1[7] = 0;
  i32x4 z4 = {0, 0, 0, 0};
  i32x8 z8 = {0, 0, 0, 0, 0, 0, 0, 0};
  __builtin_amdgcn_tensor_load_to_lds(g0, g1, z4, z4, z8, 0);
}

// ---------------------------------------------------------------------------
// Kernel 1: QKV projection.  C[8192 x 1536] = X[8192x1024] * W^T, W in {Wq,Wk,Wv}
// Block tile 128x64, 8 waves each 32x32, K-step 32, LDS staging w/ f32->bf16.
// Q,K written [L,H,N,DH]; V written transposed [L,H,DH,N].
// ---------------------------------------------------------------------------
__global__ __launch_bounds__(256)
void qkv_proj_kernel(const float* __restrict__ x,
                     const float* __restrict__ Wq,
                     const float* __restrict__ Wk,
                     const float* __restrict__ Wv,
                     unsigned short* __restrict__ qb,
                     unsigned short* __restrict__ kb,
                     unsigned short* __restrict__ vtb) {
  __shared__ unsigned short Xs[128][32];
  __shared__ unsigned short Ws[64][32];

  const int tid  = threadIdx.x;
  const int lane = tid & 31;
  const int wid  = tid >> 5;
  const int wm = wid >> 1, wn = wid & 1;
  const int m0  = blockIdx.x * 128;
  const int sel = blockIdx.y >> 3;           // 0=Q 1=K 2=V
  const int w0  = (blockIdx.y & 7) * 64;     // out-feature base within 512
  const float* W = (sel == 0) ? Wq : (sel == 1) ? Wk : Wv;
  const int kl = (lane >> 4) * 8;

  v8f c[2][2] = {};

  for (int k0 = 0; k0 < DIN; k0 += 32) {
    __syncthreads();
    {   // stage X tile 128x32 (f32 -> bf16)
      int row = tid >> 1;
      int col = (tid & 1) * 16;
      const float* src = x + (size_t)(m0 + row) * DIN + k0 + col;
      unsigned int* dst = (unsigned int*)&Xs[row][col];
#pragma unroll
      for (int i = 0; i < 8; ++i)
        dst[i] = (unsigned)f2bf(src[2*i]) | ((unsigned)f2bf(src[2*i+1]) << 16);
    }
    {   // stage W tile 64x32 (f32 -> bf16)
      int row = tid >> 2;
      int col = (tid & 3) * 8;
      const float* src = W + (size_t)(w0 + row) * DIN + k0 + col;
      unsigned int* dst = (unsigned int*)&Ws[row][col];
#pragma unroll
      for (int i = 0; i < 4; ++i)
        dst[i] = (unsigned)f2bf(src[2*i]) | ((unsigned)f2bf(src[2*i+1]) << 16);
    }
    __syncthreads();

    v16bf af[2], bfr[2];
#pragma unroll
    for (int a = 0; a < 2; ++a)
      af[a] = frag_ld_u16(&Xs[wm*32 + a*16 + (lane & 15)][kl]);
#pragma unroll
    for (int b = 0; b < 2; ++b)
      bfr[b] = frag_ld_u16(&Ws[wn*32 + b*16 + (lane & 15)][kl]);
#pragma unroll
    for (int a = 0; a < 2; ++a)
#pragma unroll
      for (int b = 0; b < 2; ++b)
        c[a][b] = wmma_bf16(af[a], bfr[b], c[a][b]);
  }

  // scatter C fragments: element r -> (M = r + 8*(lane/16), Ncol = lane%16)
#pragma unroll
  for (int a = 0; a < 2; ++a) {
#pragma unroll
    for (int b = 0; b < 2; ++b) {
#pragma unroll
      for (int r = 0; r < 8; ++r) {
        int rowm = m0 + wm*32 + a*16 + r + ((lane >> 4) << 3);
        int e    = w0 + wn*32 + b*16 + (lane & 15);
        int h = e >> 6, d = e & 63;
        int l = rowm >> 11, n = rowm & (NN - 1);
        unsigned short val = f2bf(c[a][b][r]);
        if (sel == 0)
          qb[((size_t)(l*HH + h) * NN + n) * DHH + d] = val;
        else if (sel == 1)
          kb[((size_t)(l*HH + h) * NN + n) * DHH + d] = val;
        else
          vtb[((size_t)(l*HH + h) * DHH + d) * NN + n] = val;
      }
    }
  }
}

// ---------------------------------------------------------------------------
// Kernel 2: flash attention per (l,h).  Block = 128 query rows (8 waves x 16),
// key loop in tiles of 32.  K/V^T tiles DMA'd into LDS by the Tensor Data
// Mover (wave 0 issues, s_wait_tensorcnt + barrier publishes); online softmax;
// P relayout through per-wave LDS pad; O accumulated in f32 C-fragments.
// ---------------------------------------------------------------------------
__global__ __launch_bounds__(256)
void attn_kernel(const unsigned short* __restrict__ qb,
                 const unsigned short* __restrict__ kb,
                 const unsigned short* __restrict__ vtb,
                 const unsigned char* __restrict__ mask,
                 unsigned short* __restrict__ ab) {
  __shared__ unsigned short Ks[32][64];     // keys  [j][d]
  __shared__ unsigned short Vts[64][32];    // V^T   [d][j]
  __shared__ unsigned short Ps[8][16][32];  // per-wave P pad [row][j]

  const int tid  = threadIdx.x;
  const int lane = tid & 31;
  const int w    = tid >> 5;
  const int lh   = blockIdx.x;              // l*8 + h
  const int l    = lh >> 3;
  const int h    = lh & 7;
  const int n0   = blockIdx.y * 128;
  const int kl   = (lane >> 4) * 8;

  const unsigned short* qh = qb  + (size_t)lh * NN * DHH;
  const unsigned short* kh = kb  + (size_t)lh * NN * DHH;
  const unsigned short* vh = vtb + (size_t)lh * DHH * NN;
  const unsigned char* mrow = mask + (size_t)l * NN;

  // resident Q fragments (16 rows x 64 d = two 16x32 A-frags)
  const int qrow = n0 + w*16 + (lane & 15);
  v16bf qf0 = frag_ld_u16(qh + (size_t)qrow * DHH + 0  + kl);
  v16bf qf1 = frag_ld_u16(qh + (size_t)qrow * DHH + 32 + kl);

  float m_run[8], l_run[8];
  v8f o0 = {}, o1 = {}, o2 = {}, o3 = {};
#pragma unroll
  for (int r = 0; r < 8; ++r) { m_run[r] = NEGMAX; l_run[r] = 0.f; }

  for (int j0 = 0; j0 < NN; j0 += 32) {
    __syncthreads();                   // previous tile fully consumed
    if (w == 0) {
      // K tile: 32 rows x 64 elems, row stride 64 (elems)
      tdm_load_2d_b16((unsigned)(size_t)&Ks[0][0],
                      kh + (size_t)j0 * DHH,
                      64u, (unsigned)(NN - j0), 64u, 32u, 64u);
      // V^T tile: 64 rows x 32 elems, row stride 2048 (elems)
      tdm_load_2d_b16((unsigned)(size_t)&Vts[0][0],
                      vh + j0,
                      32u, 64u, 32u, 64u, (unsigned)NN);
      __builtin_amdgcn_s_wait_tensorcnt(0);
    }
    __syncthreads();                   // DMA'd tiles visible to all waves

    // S = Q K^T : two 16-column groups, K-dim = DH = 64 -> 2 WMMAs each
    v8f s0 = {}, s1 = {};
    {
      v16bf b00 = frag_ld_u16(&Ks[ 0 + (lane & 15)][ 0 + kl]);
      v16bf b01 = frag_ld_u16(&Ks[ 0 + (lane & 15)][32 + kl]);
      s0 = wmma_bf16(qf0, b00, s0);
      s0 = wmma_bf16(qf1, b01, s0);
      v16bf b10 = frag_ld_u16(&Ks[16 + (lane & 15)][ 0 + kl]);
      v16bf b11 = frag_ld_u16(&Ks[16 + (lane & 15)][32 + kl]);
      s1 = wmma_bf16(qf0, b10, s1);
      s1 = wmma_bf16(qf1, b11, s1);
    }

    // key-padding mask: column depends only on lane
    const bool mv0 = mrow[j0 +      (lane & 15)] != 0;
    const bool mv1 = mrow[j0 + 16 + (lane & 15)] != 0;

    float t0[8], t1[8];
#pragma unroll
    for (int r = 0; r < 8; ++r) {
      t0[r] = mv0 ? s0[r] * ATTN_SCALE : NEGMAX;
      t1[r] = mv1 ? s1[r] * ATTN_SCALE : NEGMAX;
    }

    // online softmax; rows live in 16-lane halves -> xor reductions 1,2,4,8
#pragma unroll
    for (int r = 0; r < 8; ++r) {
      float rm = fmaxf(t0[r], t1[r]);
      rm = fmaxf(rm, __shfl_xor(rm, 1));
      rm = fmaxf(rm, __shfl_xor(rm, 2));
      rm = fmaxf(rm, __shfl_xor(rm, 4));
      rm = fmaxf(rm, __shfl_xor(rm, 8));
      float mn    = fmaxf(m_run[r], rm);
      float alpha = __expf(m_run[r] - mn);
      float p0 = __expf(t0[r] - mn);
      float p1 = __expf(t1[r] - mn);
      float rs = p0 + p1;
      rs += __shfl_xor(rs, 1);
      rs += __shfl_xor(rs, 2);
      rs += __shfl_xor(rs, 4);
      rs += __shfl_xor(rs, 8);
      l_run[r] = l_run[r] * alpha + rs;
      m_run[r] = mn;
      o0[r] *= alpha; o1[r] *= alpha; o2[r] *= alpha; o3[r] *= alpha;
      int prow = r + ((lane >> 4) << 3);
      Ps[w][prow][ 0 + (lane & 15)] = f2bf(p0);
      Ps[w][prow][16 + (lane & 15)] = f2bf(p1);
    }
    __syncthreads();

    // O += P * V  (K-dim = 32 keys, four 16-wide d groups)
    v16bf pf = frag_ld_u16(&Ps[w][lane & 15][kl]);
    v16bf v0 = frag_ld_u16(&Vts[ 0 + (lane & 15)][kl]);
    v16bf v1 = frag_ld_u16(&Vts[16 + (lane & 15)][kl]);
    v16bf v2 = frag_ld_u16(&Vts[32 + (lane & 15)][kl]);
    v16bf v3 = frag_ld_u16(&Vts[48 + (lane & 15)][kl]);
    o0 = wmma_bf16(pf, v0, o0);
    o1 = wmma_bf16(pf, v1, o1);
    o2 = wmma_bf16(pf, v2, o2);
    o3 = wmma_bf16(pf, v3, o3);
  }

  // normalize and write head-merged [L,N,INNER] bf16
#pragma unroll
  for (int r = 0; r < 8; ++r) {
    float inv = 1.0f / l_run[r];
    int nloc = n0 + w*16 + r + ((lane >> 4) << 3);
    size_t base = (size_t)(l * NN + nloc) * INNERD + h * DHH;
    ab[base +  0 + (lane & 15)] = f2bf(o0[r] * inv);
    ab[base + 16 + (lane & 15)] = f2bf(o1[r] * inv);
    ab[base + 32 + (lane & 15)] = f2bf(o2[r] * inv);
    ab[base + 48 + (lane & 15)] = f2bf(o3[r] * inv);
  }
}

// ---------------------------------------------------------------------------
// Kernel 3: out = attn[8192x512] * Wo^T + bo -> f32 [8192x1024]
// ---------------------------------------------------------------------------
__global__ __launch_bounds__(256)
void out_proj_kernel(const unsigned short* __restrict__ ab,
                     const float* __restrict__ Wo,
                     const float* __restrict__ bo,
                     float* __restrict__ out) {
  const int tid  = threadIdx.x;
  const int lane = tid & 31;
  const int wid  = tid >> 5;
  const int wm = wid >> 1, wn = wid & 1;
  const int m0 = blockIdx.x * 128;
  const int n0 = blockIdx.y * 64;
  const int kl = (lane >> 4) * 8;

  v8f c[2][2] = {};
  for (int k0 = 0; k0 < INNERD; k0 += 32) {
    v16bf af[2], bfr[2];
#pragma unroll
    for (int a = 0; a < 2; ++a) {
      int row = m0 + wm*32 + a*16 + (lane & 15);
      af[a] = frag_ld_u16(ab + (size_t)row * INNERD + k0 + kl);
    }
#pragma unroll
    for (int b = 0; b < 2; ++b) {
      int row = n0 + wn*32 + b*16 + (lane & 15);
      bfr[b] = frag_ld_f32(Wo + (size_t)row * INNERD + k0 + kl);
    }
#pragma unroll
    for (int a = 0; a < 2; ++a)
#pragma unroll
      for (int b = 0; b < 2; ++b)
        c[a][b] = wmma_bf16(af[a], bfr[b], c[a][b]);
  }

#pragma unroll
  for (int b = 0; b < 2; ++b) {
    int col = n0 + wn*32 + b*16 + (lane & 15);
    float bias = bo[col];
#pragma unroll
    for (int a = 0; a < 2; ++a) {
#pragma unroll
      for (int r = 0; r < 8; ++r) {
        int rowm = m0 + wm*32 + a*16 + r + ((lane >> 4) << 3);
        out[(size_t)rowm * DIN + col] = c[a][b][r] + bias;
      }
    }
  }
}

// ---------------------------------------------------------------------------
extern "C" void kernel_launch(void* const* d_in, const int* in_sizes, int n_in,
                              void* d_out, int out_size, void* d_ws, size_t ws_size,
                              hipStream_t stream) {
  (void)in_sizes; (void)n_in; (void)out_size; (void)ws_size;
  const float* x  = (const float*)d_in[0];
  const float* Wq = (const float*)d_in[1];
  const float* Wk = (const float*)d_in[2];
  const float* Wv = (const float*)d_in[3];
  const float* Wo = (const float*)d_in[4];
  const float* bo = (const float*)d_in[5];
  const unsigned char* mask = (const unsigned char*)d_in[6];

  const size_t HEADSZ = (size_t)LL * HH * NN * DHH;   // 4,194,304 elems
  unsigned short* qb  = (unsigned short*)d_ws;
  unsigned short* kb  = qb  + HEADSZ;
  unsigned short* vtb = kb  + HEADSZ;
  unsigned short* ab  = vtb + HEADSZ;

  qkv_proj_kernel<<<dim3(64, 24), 256, 0, stream>>>(x, Wq, Wk, Wv, qb, kb, vtb);
  attn_kernel   <<<dim3(32, 16), 256, 0, stream>>>(qb, kb, vtb, mask, ab);
  out_proj_kernel<<<dim3(64, 16), 256, 0, stream>>>(ab, Wo, bo, (float*)d_out);
}
